// GTN_81930796138878
// MI455X (gfx1250) — matrix operand order repaired and checked by reading, not compiled, not run on hardware
//
#include <hip/hip_runtime.h>

// ---------------- problem constants (match reference) ----------------
static constexpr int E_   = 3;
static constexpr int C_   = 2;
static constexpr int N_   = 2048;
static constexpr int B_   = 64;
static constexpr int T_   = 50;
static constexpr int WIN  = 256;
static constexpr int WOUT = 256;
static constexpr int U_   = 512;
static constexpr long NN  = (long)N_ * N_;

typedef __attribute__((ext_vector_type(16))) __bf16 v16bf;
typedef __attribute__((ext_vector_type(8)))  float  v8f;

// ===================================================================
// Generic BF16 WMMA GEMM:  C[M,N] = A[M,K] @ B[K,N]  (+ epilogue)
// A,B bf16 row-major; f32 accum via v_wmma_f32_16x16x32_bf16.
// Block tile 128x128x32, 256 threads = 8 waves arranged 4(M) x 2(N);
// wave tile 32x64 = 2x4 WMMA 16x16 tiles.
// Software-pipelined staging: next K-tile's global_load_b128 issued
// right after the barrier so the s_wait lands after the WMMA phase.
// Requirements (all call sites satisfy): K % 32 == 0, lda/ldb % 8 == 0,
// N % 128 == 0 with grid covering N exactly. Only M may be ragged.
// flags: 1=relu  2=sigmoid  4=accumulate into Cf  8=write bf16 to Cb
// ===================================================================
#define GBM 128
#define GBN 128
#define GBK 32
#define LSTR 40   // LDS row stride in bf16 elems (32 + 8 pad; 80B keeps 16B align)

__global__ __launch_bounds__(256)
void k_gemm_bf16(const __bf16* __restrict__ A, const __bf16* __restrict__ B,
                 float* __restrict__ Cf, __bf16* __restrict__ Cb,
                 int M, int N, int K, int lda, int ldb, int ldc,
                 long sAz, long sBz, long sCz,
                 const float* __restrict__ bias, int flags)
{
    __shared__ __bf16 As[GBM * LSTR];   // As[m][k]
    __shared__ __bf16 Bs[GBN * LSTR];   // Bs[n][k]  (transposed on store)

    const int z = blockIdx.z;
    A += (long)z * sAz;
    B += (long)z * sBz;
    if (Cf) Cf += (long)z * sCz;
    if (Cb) Cb += (long)z * sCz;

    const int tid  = threadIdx.x;
    const int wave = tid >> 5;
    const int lane = tid & 31;
    const int half = lane >> 4;
    const int lrow = lane & 15;

    const int bm0 = blockIdx.x * GBM;
    const int bn0 = blockIdx.y * GBN;
    const int wm0 = (wave >> 1) * 32;   // 4 wave rows
    const int wn0 = (wave & 1) * 64;    // 2 wave cols

    // staging coordinates (branch-free; A rows clamped: OOB rows read valid
    // memory and only feed accumulator rows that are never stored)
    const int amQ = (tid >> 2);          // A row of quad (it=1 adds 64)
    const int akQ = (tid & 3) * 8;       // A k-offset of 8-elem quad
    const int bkQ = (tid >> 4);          // B k row of quad (it=1 adds 16)
    const int bnQ = (tid & 15) * 8;      // B n-offset of 8-elem quad

    int gmA0 = bm0 + amQ;      if (gmA0 > M - 1) gmA0 = M - 1;
    int gmA1 = bm0 + amQ + 64; if (gmA1 > M - 1) gmA1 = M - 1;

    const __bf16* pA0 = A + (long)gmA0 * lda + akQ;
    const __bf16* pA1 = A + (long)gmA1 * lda + akQ;
    const __bf16* pB0 = B + (long)bkQ * ldb + bn0 + bnQ;
    const __bf16* pB1 = B + (long)(bkQ + 16) * ldb + bn0 + bnQ;

    v8f acc[2][4];
    #pragma unroll
    for (int a = 0; a < 2; ++a)
        #pragma unroll
        for (int b = 0; b < 4; ++b)
            #pragma unroll
            for (int r = 0; r < 8; ++r) acc[a][b][r] = 0.0f;

    union AF { v16bf v; unsigned u[8]; };
    union Q  { uint4 q; __bf16 h[8]; };

    // prologue: load K-tile 0 into registers
    uint4 ra0 = *(const uint4*)pA0;
    uint4 ra1 = *(const uint4*)pA1;
    uint4 rb0 = *(const uint4*)pB0;
    uint4 rb1 = *(const uint4*)pB1;

    for (int k0 = 0; k0 < K; k0 += GBK) {
        // --- commit staged registers to LDS ---
        *(uint4*)&As[amQ * LSTR + akQ]        = ra0;
        *(uint4*)&As[(amQ + 64) * LSTR + akQ] = ra1;
        {
            Q d0; d0.q = rb0;
            Q d1; d1.q = rb1;
            #pragma unroll
            for (int j = 0; j < 8; ++j) {
                Bs[(bnQ + j) * LSTR + bkQ]      = d0.h[j];
                Bs[(bnQ + j) * LSTR + bkQ + 16] = d1.h[j];
            }
        }
        __syncthreads();

        // --- issue next K-tile's global loads (latency hidden by WMMAs) ---
        int kn = k0 + GBK;
        if (kn < K) {
            if (kn + GBK < K) {
                __builtin_prefetch(pA0 + kn + GBK, 0, 1);
                __builtin_prefetch(pB0 + (long)(kn + GBK) * ldb, 0, 1);
            }
            ra0 = *(const uint4*)(pA0 + kn);
            ra1 = *(const uint4*)(pA1 + kn);
            rb0 = *(const uint4*)(pB0 + (long)kn * ldb);
            rb1 = *(const uint4*)(pB1 + (long)kn * ldb);
        }

        // --- load fragments per CDNA5 VGPR layouts (all ds_load_b128) ---
        AF af[2], bf[4];
        #pragma unroll
        for (int tm = 0; tm < 2; ++tm) {
            int row = wm0 + tm * 16 + lrow;
            #pragma unroll
            for (int v = 0; v < 8; ++v) {
                // A 16x32 bf16: v<4 -> K = 2v + 8*half ; v>=4 -> K = 8 + 2v + 8*half
                int kk = (v < 4) ? (2 * v + 8 * half) : (8 + 2 * v + 8 * half);
                af[tm].u[v] = *(const unsigned*)&As[row * LSTR + kk];
            }
        }
        #pragma unroll
        for (int tn = 0; tn < 4; ++tn) {
            int col = wn0 + tn * 16 + lrow;
            int kb  = 8 * half;             // v<4 -> K = 2v + 8*half ; v>=4 -> K = 8 + 2v + 8*half
            #pragma unroll
            for (int v = 0; v < 8; ++v)
                bf[tn].u[v] = *(const unsigned*)&Bs[col * LSTR + kb + 2 * v + (v < 4 ? 0 : 8)];
        }

        #pragma unroll
        for (int tm = 0; tm < 2; ++tm)
            #pragma unroll
            for (int tn = 0; tn < 4; ++tn)
                acc[tm][tn] = __builtin_amdgcn_wmma_f32_16x16x32_bf16(
                    false, af[tm].v, false, bf[tn].v,
                    (short)0, acc[tm][tn], false, false);
        __syncthreads();
    }

    // --- epilogue: D VGPR r -> M = r + 8*half, N = lane%16 ---
    #pragma unroll
    for (int tm = 0; tm < 2; ++tm)
    for (int tn = 0; tn < 4; ++tn)
    #pragma unroll
    for (int r = 0; r < 8; ++r) {
        int m = bm0 + wm0 + tm * 16 + r + 8 * half;
        int n = bn0 + wn0 + tn * 16 + lrow;
        if (m < M) {
            float v = acc[tm][tn][r];
            if (bias) v += bias[n];
            if (flags & 1) v = fmaxf(v, 0.0f);
            if (flags & 2) v = 1.0f / (1.0f + __expf(-v));
            long off = (long)m * ldc + n;
            if (flags & 4) v += Cf[off];
            if (flags & 8) Cb[off] = (__bf16)v;
            else           Cf[off] = v;
        }
    }
}

// ---------------- GT conv: softmax over edge types + combine --------
__device__ __forceinline__ void softmax3(const float* w, float* f) {
    float m  = fmaxf(w[0], fmaxf(w[1], w[2]));
    float e0 = __expf(w[0] - m), e1 = __expf(w[1] - m), e2 = __expf(w[2] - m);
    float s  = e0 + e1 + e2;
    f[0] = e0 / s; f[1] = e1 / s; f[2] = e2 / s;
}

__global__ void k_combine(const float* __restrict__ A,
                          const float* __restrict__ w1,
                          const float* __restrict__ w2,
                          const float* __restrict__ w3,
                          __bf16* __restrict__ HA, __bf16* __restrict__ HB,
                          __bf16* __restrict__ HB2)
{
    long idx = (long)blockIdx.x * 256 + threadIdx.x;
    int  c   = blockIdx.y;
    if (idx >= NN) return;
    float f1[3], f2[3], f3[3];
    softmax3(w1 + c * E_, f1);
    softmax3(w2 + c * E_, f2);
    softmax3(w3 + c * E_, f3);
    float a0 = A[idx], a1 = A[NN + idx], a2 = A[2 * NN + idx];
    long o = (long)c * NN + idx;
    HA [o] = (__bf16)(f1[0] * a0 + f1[1] * a1 + f1[2] * a2);
    HB [o] = (__bf16)(f2[0] * a0 + f2[1] * a1 + f2[2] * a2);
    HB2[o] = (__bf16)(f3[0] * a0 + f3[1] * a1 + f3[2] * a2);
}

// --------- column sums (normalization degrees) ----------------------
__global__ void k_colnorm_deg(const float* __restrict__ H, float* __restrict__ dinv)
{
    int m = blockIdx.x * 256 + threadIdx.x;
    int c = blockIdx.y;
    if (m >= N_) return;
    const float* Hc = H + (long)c * NN;
    float s = 0.0f;
    for (int n = 0; n < N_; ++n)
        if (n != m) s += Hc[(long)n * N_ + m];
    dinv[c * N_ + m] = (s > 0.0f) ? (1.0f / s) : 0.0f;
}

__global__ void k_colnorm_apply(const float* __restrict__ H,
                                const float* __restrict__ dinv,
                                __bf16* __restrict__ Hn)
{
    long idx = (long)blockIdx.x * 256 + threadIdx.x;
    int  c   = blockIdx.y;
    if (idx >= NN) return;
    int n = (int)(idx / N_), m = (int)(idx % N_);
    float v = (n == m) ? 0.0f : H[(long)c * NN + idx] * dinv[c * N_ + m];
    Hn[(long)c * NN + idx] = (__bf16)v;
}

__global__ void k_gcn_deg(const float* __restrict__ H2, float* __restrict__ dis)
{
    int m = blockIdx.x * 256 + threadIdx.x;
    int c = blockIdx.y;
    if (m >= N_) return;
    const float* Hc = H2 + (long)c * NN;
    float s = 1.0f;                 // + self loop
    for (int n = 0; n < N_; ++n) s += Hc[(long)n * N_ + m];
    dis[c * N_ + m] = (s > 0.0f) ? rsqrtf(s) : 0.0f;
}

// AhatT[c][t][s] = (H2[c][s][t] + (s==t)) * dis[s]*dis[t]   (tiled transpose)
__global__ void k_ahatT(const float* __restrict__ H2, const float* __restrict__ dis,
                        __bf16* __restrict__ AT)
{
    __shared__ float tile[32][33];
    int c  = blockIdx.z;
    int s0 = blockIdx.y * 32, t0 = blockIdx.x * 32;
    int tx = threadIdx.x & 31, ty = threadIdx.x >> 5;
    const float* dc = dis + c * N_;
    for (int i = ty; i < 32; i += 8) {
        int s = s0 + i, t = t0 + tx;
        float v = H2[((long)c * N_ + s) * N_ + t];
        if (s == t) v += 1.0f;
        tile[i][tx] = v * dc[s] * dc[t];
    }
    __syncthreads();
    for (int i = ty; i < 32; i += 8) {
        int t = t0 + i, s = s0 + tx;
        AT[((long)c * N_ + t) * N_ + s] = (__bf16)tile[tx][i];
    }
}

// ---------------- converts ------------------------------------------
__global__ void k_cvt_bf16(const float* __restrict__ in, __bf16* __restrict__ out, long n)
{
    for (long i = (long)blockIdx.x * 256 + threadIdx.x; i < n; i += (long)gridDim.x * 256)
        out[i] = (__bf16)in[i];
}

// out[c][r] = in[r][c] ; R,C multiples of 32
__global__ void k_transpose_cvt(const float* __restrict__ in, __bf16* __restrict__ out,
                                int R, int C)
{
    __shared__ float tile[32][33];
    int r0 = blockIdx.y * 32, c0 = blockIdx.x * 32;
    int tx = threadIdx.x & 31, ty = threadIdx.x >> 5;
    for (int i = ty; i < 32; i += 8)
        tile[i][tx] = in[(long)(r0 + i) * C + c0 + tx];
    __syncthreads();
    for (int i = ty; i < 32; i += 8)
        out[(long)(c0 + i) * R + r0 + tx] = (__bf16)tile[tx][i];
}

// ---------------- basket max pooling --------------------------------
__global__ __launch_bounds__(256)
void k_pool(const float* __restrict__ seqs, const float* __restrict__ X_,
            float* __restrict__ bs)
{
    __shared__ int s_idx[N_];
    __shared__ int s_cnt;
    int b = blockIdx.x, t = blockIdx.y;
    if (threadIdx.x == 0) s_cnt = 0;
    __syncthreads();
    const float* mask = seqs + ((long)b * T_ + t) * N_;
    for (int n = threadIdx.x; n < N_; n += 256)
        if (mask[n] > 0.0f) { int p = atomicAdd(&s_cnt, 1); s_idx[p] = n; }
    __syncthreads();
    int cnt = s_cnt;
    int j = threadIdx.x;               // WOUT == 256 == blockDim
    float m = -3.4e38f;
    for (int i = 0; i < cnt; ++i)
        m = fmaxf(m, X_[(long)s_idx[i] * WOUT + j]);
    bs[((long)t * B_ + b) * WOUT + j] = (cnt > 0) ? m : 0.0f;
}

// ---------------- LSTM cell -----------------------------------------
__global__ void k_lstm_cell(const float* __restrict__ gates,
                            const float* __restrict__ bih,
                            const float* __restrict__ bhh,
                            float* __restrict__ h, float* __restrict__ c,
                            float* __restrict__ outs_t)
{
    int idx = blockIdx.x * 256 + threadIdx.x;
    if (idx >= B_ * U_) return;
    int b = idx / U_, u = idx % U_;
    const float* g = gates + (long)b * 4 * U_;
    float gi = g[u]            + bih[u]            + bhh[u];
    float gf = g[U_ + u]       + bih[U_ + u]       + bhh[U_ + u];
    float gg = g[2 * U_ + u]   + bih[2 * U_ + u]   + bhh[2 * U_ + u];
    float go = g[3 * U_ + u]   + bih[3 * U_ + u]   + bhh[3 * U_ + u];
    float si = 1.0f / (1.0f + __expf(-gi));
    float sf = 1.0f / (1.0f + __expf(-gf));
    float so = 1.0f / (1.0f + __expf(-go));
    float cc = sf * c[idx] + si * tanhf(gg);
    float hh = so * tanhf(cc);
    c[idx] = cc; h[idx] = hh; outs_t[idx] = hh;
}

// ---------------- gather last valid step ----------------------------
__global__ void k_gather(const float* __restrict__ outs, const int* __restrict__ seq_len,
                         __bf16* __restrict__ actual)
{
    int idx = blockIdx.x * 256 + threadIdx.x;
    if (idx >= B_ * U_) return;
    int b = idx / U_, u = idx % U_;
    int t = seq_len[b] - 1;
    actual[idx] = (__bf16)outs[((long)t * B_ + b) * U_ + u];
}

// ===================================================================
extern "C" void kernel_launch(void* const* d_in, const int* in_sizes, int n_in,
                              void* d_out, int out_size, void* d_ws, size_t ws_size,
                              hipStream_t stream)
{
    (void)in_sizes; (void)n_in; (void)out_size; (void)ws_size;

    const float* A_      = (const float*)d_in[0];
    const float* X       = (const float*)d_in[1];
    const float* seqs    = (const float*)d_in[2];
    const int*   seq_len = (const int*)  d_in[3];
    const float* h0      = (const float*)d_in[4];
    const float* c0      = (const float*)d_in[5];
    const float* gt_w1   = (const float*)d_in[6];
    const float* gt_w2   = (const float*)d_in[7];
    const float* gt_w3   = (const float*)d_in[8];
    const float* gcn_W   = (const float*)d_in[9];
    const float* gcn_b   = (const float*)d_in[10];
    const float* lin1_W  = (const float*)d_in[11];
    const float* lin1_b  = (const float*)d_in[12];
    const float* Wih     = (const float*)d_in[13];
    const float* Whh     = (const float*)d_in[14];
    const float* bih     = (const float*)d_in[15];
    const float* bhh     = (const float*)d_in[16];
    const float* score_W = (const float*)d_in[17];
    float* out = (float*)d_out;

    // ------------- workspace bump allocator -------------
    char* p = (char*)d_ws;
    auto alloc = [&](size_t bytes) {
        char* r = p;
        p += (bytes + 255) & ~(size_t)255;
        return r;
    };
    __bf16* HA    = (__bf16*)alloc((size_t)C_ * NN * 2);
    __bf16* HB    = (__bf16*)alloc((size_t)C_ * NN * 2);
    __bf16* HB2   = (__bf16*)alloc((size_t)C_ * NN * 2);
    float*  Hbuf  = (float*) alloc((size_t)C_ * NN * 4);   // H, then H2
    float*  dinv  = (float*) alloc((size_t)C_ * N_ * 4);
    float*  dis   = (float*) alloc((size_t)C_ * N_ * 4);
    __bf16* Xbf   = (__bf16*)alloc((size_t)N_ * WIN * 2);
    __bf16* gcnWb = (__bf16*)alloc((size_t)WIN * WOUT * 2);
    __bf16* XWbf  = (__bf16*)alloc((size_t)N_ * WOUT * 2);
    __bf16* Xcat  = (__bf16*)alloc((size_t)N_ * C_ * WOUT * 2);
    __bf16* lin1Wb= (__bf16*)alloc((size_t)C_ * WOUT * WOUT * 2);
    float*  Xout  = (float*) alloc((size_t)N_ * WOUT * 4);
    float*  bs    = (float*) alloc((size_t)T_ * B_ * WOUT * 4);
    __bf16* WihT  = (__bf16*)alloc((size_t)WOUT * 4 * U_ * 2);
    __bf16* WhhT  = (__bf16*)alloc((size_t)U_ * 4 * U_ * 2);
    __bf16* scWT  = (__bf16*)alloc((size_t)U_ * N_ * 2);
    float*  gates = (float*) alloc((size_t)B_ * 4 * U_ * 4);
    float*  hbuf  = (float*) alloc((size_t)B_ * U_ * 4);
    float*  cbuf  = (float*) alloc((size_t)B_ * U_ * 4);
    __bf16* xbf   = (__bf16*)alloc((size_t)B_ * WOUT * 2);
    __bf16* hbf   = (__bf16*)alloc((size_t)B_ * U_ * 2);
    float*  outs  = (float*) alloc((size_t)T_ * B_ * U_ * 4);
    __bf16* actbf = (__bf16*)alloc((size_t)B_ * U_ * 2);
    // aliases (dead buffers reused)
    __bf16* Hn  = HA;    // HA dead after GEMM1
    __bf16* AT  = HB;    // HB dead after GEMM1
    float*  H2  = Hbuf;  // H dead after colnorm_apply

    const dim3 blk(256);

    // 1) GT conv combines -> HA, HB, HB2 (bf16)
    k_combine<<<dim3((unsigned)(NN / 256), C_), blk, 0, stream>>>(
        A_, gt_w1, gt_w2, gt_w3, HA, HB, HB2);

    // 2) H = HA @ HB   (per channel, 2048^3 bf16 WMMA, f32 out)
    k_gemm_bf16<<<dim3(16, 16, C_), blk, 0, stream>>>(
        HA, HB, Hbuf, nullptr, N_, N_, N_, N_, N_, N_, NN, NN, NN, nullptr, 0);

    // 3) column-normalize (no diag) -> Hn (bf16)
    k_colnorm_deg  <<<dim3(N_ / 256, C_), blk, 0, stream>>>(Hbuf, dinv);
    k_colnorm_apply<<<dim3((unsigned)(NN / 256), C_), blk, 0, stream>>>(Hbuf, dinv, Hn);

    // 4) H2 = Hn @ HB2
    k_gemm_bf16<<<dim3(16, 16, C_), blk, 0, stream>>>(
        Hn, HB2, H2, nullptr, N_, N_, N_, N_, N_, N_, NN, NN, NN, nullptr, 0);

    // 5) GCN: sym-normalized transposed adjacency (bf16)
    k_gcn_deg<<<dim3(N_ / 256, C_), blk, 0, stream>>>(H2, dis);
    k_ahatT  <<<dim3(N_ / 32, N_ / 32, C_), blk, 0, stream>>>(H2, dis, AT);

    // 6) XW = X @ gcn_W  (bf16 out, no bias yet)
    k_cvt_bf16<<<dim3(1024), blk, 0, stream>>>(X, Xbf, (long)N_ * WIN);
    k_cvt_bf16<<<dim3(256),  blk, 0, stream>>>(gcn_W, gcnWb, (long)WIN * WOUT);
    k_gemm_bf16<<<dim3(16, 2, 1), blk, 0, stream>>>(
        Xbf, gcnWb, nullptr, XWbf, N_, WOUT, WIN, WIN, WOUT, WOUT,
        0, 0, 0, nullptr, 8);

    // 7) feats[c] = relu(AT[c] @ XW + gcn_b) -> Xcat[:, c*256:(c+1)*256] (bf16)
    k_gemm_bf16<<<dim3(16, 2, C_), blk, 0, stream>>>(
        AT, XWbf, nullptr, Xcat, N_, WOUT, N_, N_, WOUT, C_ * WOUT,
        NN, 0, WOUT, gcn_b, 1 | 8);

    // 8) X_ = Xcat @ lin1_W + lin1_b  (f32)
    k_cvt_bf16<<<dim3(512), blk, 0, stream>>>(lin1_W, lin1Wb, (long)C_ * WOUT * WOUT);
    k_gemm_bf16<<<dim3(16, 2, 1), blk, 0, stream>>>(
        Xcat, lin1Wb, Xout, nullptr, N_, WOUT, C_ * WOUT, C_ * WOUT, WOUT, WOUT,
        0, 0, 0, lin1_b, 0);

    // 9) basket max pooling -> bs[T,B,WOUT]
    k_pool<<<dim3(B_, T_), blk, 0, stream>>>(seqs, Xout, bs);

    // 10) LSTM weights transposed to [K, 4U] bf16
    k_transpose_cvt<<<dim3(WOUT / 32, 4 * U_ / 32), blk, 0, stream>>>(Wih, WihT, 4 * U_, WOUT);
    k_transpose_cvt<<<dim3(U_ / 32,   4 * U_ / 32), blk, 0, stream>>>(Whh, WhhT, 4 * U_, U_);
    k_transpose_cvt<<<dim3(U_ / 32,   N_ / 32),     blk, 0, stream>>>(score_W, scWT, N_, U_);

    hipMemcpyAsync(hbuf, h0, (size_t)B_ * U_ * 4, hipMemcpyDeviceToDevice, stream);
    hipMemcpyAsync(cbuf, c0, (size_t)B_ * U_ * 4, hipMemcpyDeviceToDevice, stream);

    // 11) 50 LSTM steps
    for (int t = 0; t < T_; ++t) {
        k_cvt_bf16<<<dim3(64),  blk, 0, stream>>>(bs + (long)t * B_ * WOUT, xbf, (long)B_ * WOUT);
        k_cvt_bf16<<<dim3(128), blk, 0, stream>>>(hbuf, hbf, (long)B_ * U_);
        // gates = x @ WihT
        k_gemm_bf16<<<dim3(1, 16, 1), blk, 0, stream>>>(
            xbf, WihT, gates, nullptr, B_, 4 * U_, WOUT, WOUT, 4 * U_, 4 * U_,
            0, 0, 0, nullptr, 0);
        // gates += h @ WhhT
        k_gemm_bf16<<<dim3(1, 16, 1), blk, 0, stream>>>(
            hbf, WhhT, gates, nullptr, B_, 4 * U_, U_, U_, 4 * U_, 4 * U_,
            0, 0, 0, nullptr, 4);
        k_lstm_cell<<<dim3((B_ * U_) / 256), blk, 0, stream>>>(
            gates, bih, bhh, hbuf, cbuf, outs + (long)t * B_ * U_);
    }

    // 12) gather last valid step, score + sigmoid -> d_out [B,N]
    k_gather<<<dim3((B_ * U_) / 256), blk, 0, stream>>>(outs, seq_len, actbf);
    k_gemm_bf16<<<dim3(1, 16, 1), blk, 0, stream>>>(
        actbf, scWT, out, nullptr, B_, N_, U_, U_, N_, N_,
        0, 0, 0, nullptr, 2);
}
